// Qwen3_5GatedDeltaNet_4853313044757
// MI455X (gfx1250) — compile-verified, matrix operand order checked by reading, not run on hardware
//
#include <hip/hip_runtime.h>
#include <stddef.h>
#include <stdint.h>

// ---------------------------------------------------------------------------
// Problem constants (from the reference)
// ---------------------------------------------------------------------------
#define H_SIZE     2048
#define NUM_K_HEADS 16
#define NUM_V_HEADS 32
#define DK         128
#define DV         128
#define KEY_DIM    (NUM_K_HEADS * DK)          // 2048
#define VALUE_DIM  (NUM_V_HEADS * DV)          // 4096
#define CONV_DIM   (2 * KEY_DIM + VALUE_DIM)   // 8192
#define KERNEL_SZ  4
#define BB         2
#define SS         2048
#define BS         (BB * SS)                   // 4096 rows

typedef __attribute__((ext_vector_type(16))) __bf16   v16bf;
typedef __attribute__((ext_vector_type(8)))  __bf16   v8bf;
typedef __attribute__((ext_vector_type(8)))  float    v8f;
typedef __attribute__((ext_vector_type(4)))  uint32_t v4u;
typedef __attribute__((ext_vector_type(4)))  int      v4i;
typedef __attribute__((ext_vector_type(8)))  int      v8i;

// ---------------------------------------------------------------------------
// fp32 -> bf16 convert (plain)
// ---------------------------------------------------------------------------
__global__ __launch_bounds__(256) void cvt_f32_bf16(const float* __restrict__ in,
                                                    __bf16* __restrict__ out, int n) {
  int i = blockIdx.x * 256 + threadIdx.x;
  if (i < n) out[i] = (__bf16)in[i];
}

// ---------------------------------------------------------------------------
// fp32 (K x N) -> bf16 transposed (N x K): weights become contiguous along K
// so the Tensor Data Mover can stage B tiles with a simple 2D descriptor.
// ---------------------------------------------------------------------------
__global__ __launch_bounds__(256) void cvt_transpose_bf16(
    const float* __restrict__ in, __bf16* __restrict__ out, int K, int N) {
  __shared__ float t[32][33];
  const int n0 = blockIdx.x * 32;
  const int k0 = blockIdx.y * 32;
  for (int r = threadIdx.y; r < 32; r += 8)
    t[r][threadIdx.x] = in[(size_t)(k0 + r) * N + n0 + threadIdx.x];
  __syncthreads();
  for (int r = threadIdx.y; r < 32; r += 8)
    out[(size_t)(n0 + r) * K + k0 + threadIdx.x] = (__bf16)t[threadIdx.x][r];
}

// ---------------------------------------------------------------------------
// Tensor DMA descriptor helpers (CDNA5 ISA ch.8, D# groups 0/1).
// 2D tensor, data_size = 2 bytes, tile = tile_rows x 32 elements, with LDS
// padding (interval 16 dwords = 64B row, amount 4 dwords = 16B) so the DMA
// writes rows at an 80-byte stride matching the [..][40] LDS arrays.
// ---------------------------------------------------------------------------
__device__ __forceinline__ v4u tdm_group0(uint32_t lds_off, const void* gptr) {
  const uint64_t ga = (uint64_t)(uintptr_t)gptr;
  v4u g0;
  g0[0] = 1u;                                   // count=1, user descriptor
  g0[1] = lds_off;                              // LDS byte address
  g0[2] = (uint32_t)(ga & 0xffffffffu);         // global_addr[31:0]
  g0[3] = (uint32_t)((ga >> 32) & 0x1ffffffu)   // global_addr[56:32]
          | (2u << 30);                         // type = 2 ("image")
  return g0;
}

__device__ __forceinline__ v8i tdm_group1(uint32_t dim0, uint32_t dim1,
                                          uint32_t tile_rows) {
  // fields: data_size=1 (2B) @16, pad_enable @20, pad_interval=3 (16 dw) @22,
  // pad_amount=3 (4 dw) @25; tensor_dim0 @48, tensor_dim1 @80, tile_dim0 @112,
  // tile_dim1 @128, tensor_dim0_stride @160 (= dim0, packed rows)
  v8i g1;
  g1[0] = (int)((1u << 16) | (1u << 20) | (3u << 22) | (3u << 25));
  g1[1] = (int)((dim0 & 0xffffu) << 16);
  g1[2] = (int)((dim0 >> 16) | ((dim1 & 0xffffu) << 16));
  g1[3] = (int)((dim1 >> 16) | (32u << 16));            // tile_dim0 = 32
  g1[4] = (int)(tile_rows & 0xffffu);                    // tile_dim1
  g1[5] = (int)dim0;                                     // stride0[31:0]
  g1[6] = 0;                                             // stride0 hi / stride1 lo
  g1[7] = 0;
  return g1;
}

__device__ __forceinline__ void tdm_load(v4u g0, v8i g1) {
  const v4i z4 = {0, 0, 0, 0};
#if defined(__clang_major__) && __clang_major__ >= 23
  const v8i z8 = {0, 0, 0, 0, 0, 0, 0, 0};
  __builtin_amdgcn_tensor_load_to_lds(g0, g1, z4, z4, z8, 0);
#else
  __builtin_amdgcn_tensor_load_to_lds(g0, g1, z4, z4, 0);
#endif
}

// ---------------------------------------------------------------------------
// TDM-fed WMMA GEMM:  C[M,N] = A[M,K] * Bt[N,K]^T   (A, Bt bf16 row-major,
// both contiguous along K; fp32 accumulate).
// Block = 256 threads (8 waves), block tile 128x128, wave tile 32x64
// (2x4 of 16x16x32 WMMA). K step 32, double-buffered LDS filled by the
// Tensor Data Mover (wave 0 issues, TENSORcnt-synchronized).
// ---------------------------------------------------------------------------
__global__ __launch_bounds__(256) void gemm_bf16_wmma_tdm(
    const __bf16* __restrict__ A, const __bf16* __restrict__ Bt,
    void* __restrict__ Cout, int M, int N, int K, int bf16out) {
  __shared__ __align__(16) __bf16 As[2][128][40];  // 80B row stride (TDM pad)
  __shared__ __align__(16) __bf16 Bs[2][128][40];

  const int tid  = threadIdx.x;
  const int lane = tid & 31;
  const int w    = tid >> 5;
  const int wm   = w >> 1;            // 0..3  (M direction, 32 rows each)
  const int wn   = w & 1;             // 0..1  (N direction, 64 cols each)
  const int mb   = blockIdx.y * 128;
  const int nb   = blockIdx.x * 128;
  const int lr   = lane & 15;
  const int kbase = (lane < 16) ? 0 : 8;   // ISA 16-bit operand lane K split
  const bool issuer = (w == 0);

  v8f c[2][4] = {};

  if (issuer) {  // prologue: DMA tile kb=0 into buffer 0
    tdm_load(tdm_group0((uint32_t)(uintptr_t)&As[0][0][0],
                        A + (size_t)mb * K),
             tdm_group1((uint32_t)K, (uint32_t)M, 128));
    tdm_load(tdm_group0((uint32_t)(uintptr_t)&Bs[0][0][0],
                        Bt + (size_t)nb * K),
             tdm_group1((uint32_t)K, (uint32_t)N, 128));
  }

  int cur = 0;
  for (int kb = 0; kb < K; kb += 32) {
    if (issuer) {
      if (kb + 32 < K) {  // DMA next tile into the other buffer, then wait
        tdm_load(tdm_group0((uint32_t)(uintptr_t)&As[cur ^ 1][0][0],
                            A + (size_t)mb * K + (kb + 32)),
                 tdm_group1((uint32_t)K, (uint32_t)M, 128));
        tdm_load(tdm_group0((uint32_t)(uintptr_t)&Bs[cur ^ 1][0][0],
                            Bt + (size_t)nb * K + (kb + 32)),
                 tdm_group1((uint32_t)K, (uint32_t)N, 128));
        __builtin_amdgcn_s_wait_tensorcnt(2);   // current tile landed
      } else {
        __builtin_amdgcn_s_wait_tensorcnt(0);
      }
    }
    __syncthreads();   // tile `cur` visible to all waves

    v16bf afr[2], bfr[4];
#pragma unroll
    for (int i = 0; i < 2; ++i) {
      const __bf16* ap = &As[cur][wm * 32 + i * 16 + lr][kbase];
      v8bf lo = *(const v8bf*)ap;
      v8bf hi = *(const v8bf*)(ap + 16);
      afr[i] = __builtin_shufflevector(lo, hi, 0, 1, 2, 3, 4, 5, 6, 7,
                                       8, 9, 10, 11, 12, 13, 14, 15);
    }
#pragma unroll
    for (int j = 0; j < 4; ++j) {
      const __bf16* bp = &Bs[cur][wn * 64 + j * 16 + lr][kbase];
      v8bf lo = *(const v8bf*)bp;
      v8bf hi = *(const v8bf*)(bp + 16);
      bfr[j] = __builtin_shufflevector(lo, hi, 0, 1, 2, 3, 4, 5, 6, 7,
                                       8, 9, 10, 11, 12, 13, 14, 15);
    }
#pragma unroll
    for (int i = 0; i < 2; ++i)
#pragma unroll
      for (int j = 0; j < 4; ++j)
        c[i][j] = __builtin_amdgcn_wmma_f32_16x16x32_bf16(
            false, afr[i], false, bfr[j], (short)0, c[i][j], false, false);

    __syncthreads();   // all waves done reading buffer `cur` before reuse
    cur ^= 1;
  }

  // Store C per ISA 16x16 f32 layout: VGPR r -> (M=r | r+8, N=lane&15)
  const int rofs = (lane >= 16) ? 8 : 0;
#pragma unroll
  for (int i = 0; i < 2; ++i)
#pragma unroll
    for (int j = 0; j < 4; ++j)
#pragma unroll
      for (int r = 0; r < 8; ++r) {
        const int mm = mb + wm * 32 + i * 16 + rofs + r;
        const int nn = nb + wn * 64 + j * 16 + lr;
        const float v = c[i][j][r];
        if (bf16out)
          ((__bf16*)Cout)[(size_t)mm * N + nn] = (__bf16)v;
        else
          ((float*)Cout)[(size_t)mm * N + nn] = v;
      }
}

// ---------------------------------------------------------------------------
// a/b head projections + gate math:
//   b = hs @ w_b ; a = hs @ w_a ; beta = sigmoid(b)
//   g = -exp(A_log) * softplus(a + dt_bias)
// ---------------------------------------------------------------------------
__global__ __launch_bounds__(256) void ab_gate_kernel(
    const float* __restrict__ hs, const float* __restrict__ w_b,
    const float* __restrict__ w_a, const float* __restrict__ dt_bias,
    const float* __restrict__ A_log, float* __restrict__ beta,
    float* __restrict__ g) {
  const int bs  = blockIdx.x;
  const int tid = threadIdx.x;
  const int h   = tid >> 3;     // 0..31
  const int p   = tid & 7;      // 0..7
  const float* row = hs + (size_t)bs * H_SIZE;

  float sb = 0.f, sa = 0.f;
  for (int i = p * 256; i < (p + 1) * 256; ++i) {
    const float x = row[i];
    sb += x * w_b[(size_t)i * NUM_V_HEADS + h];
    sa += x * w_a[(size_t)i * NUM_V_HEADS + h];
  }
  __shared__ float rb[256], ra[256];
  rb[tid] = sb; ra[tid] = sa;
  __syncthreads();
  if (p == 0) {
    float tb = 0.f, ta = 0.f;
#pragma unroll
    for (int q = 0; q < 8; ++q) { tb += rb[h * 8 + q]; ta += ra[h * 8 + q]; }
    const float x = ta + dt_bias[h];
    const float sp = (x > 20.f) ? x : log1pf(expf(x));
    beta[(size_t)bs * NUM_V_HEADS + h] = 1.f / (1.f + expf(-tb));
    g[(size_t)bs * NUM_V_HEADS + h]    = -expf(A_log[h]) * sp;
  }
}

// ---------------------------------------------------------------------------
// Causal depthwise conv (K=4) + SiLU, then split into q (l2norm * DK^-0.5),
// k (l2norm), v. One block = one head's 128 channels at one (b,s).
// ---------------------------------------------------------------------------
__global__ __launch_bounds__(128) void conv_silu_split_kernel(
    const __bf16* __restrict__ mixed, const float* __restrict__ conv_w,
    float* __restrict__ q, float* __restrict__ k, float* __restrict__ v) {
  const int cb  = blockIdx.x;        // 0..63
  const int bs  = blockIdx.y;        // 0..B*S-1
  const int b   = bs / SS;
  const int s   = bs % SS;
  const int tid = threadIdx.x;
  const int ch  = cb * 128 + tid;

  float acc = 0.f;
#pragma unroll
  for (int j = 0; j < KERNEL_SZ; ++j) {
    const int ts = s - (KERNEL_SZ - 1) + j;
    if (ts >= 0) {
      const float x = (float)mixed[((size_t)(b * SS + ts)) * CONV_DIM + ch];
      acc += x * conv_w[(size_t)ch * KERNEL_SZ + j];
    }
  }
  const float y = acc / (1.f + expf(-acc));   // silu

  if (cb < 32) {
    __shared__ float red[128];
    red[tid] = y * y;
    __syncthreads();
    for (int st = 64; st > 0; st >>= 1) {
      if (tid < st) red[tid] += red[tid + st];
      __syncthreads();
    }
    const float inv = rsqrtf(red[0] + 1e-6f);
    if (cb < 16) {
      q[(size_t)bs * KEY_DIM + ch] = y * inv * 0.08838834764831845f;  // DK^-0.5
    } else {
      k[(size_t)bs * KEY_DIM + (ch - KEY_DIM)] = y * inv;
    }
  } else {
    v[(size_t)bs * VALUE_DIM + (ch - 2 * KEY_DIM)] = y;
  }
}

// ---------------------------------------------------------------------------
// Gated delta-rule recurrence. One block per (b, v-head); 128 threads; each
// thread owns one DV column of the 128x128 state, held in registers.
// ---------------------------------------------------------------------------
__global__ __launch_bounds__(128) void deltanet_recurrence_kernel(
    const float* __restrict__ q, const float* __restrict__ k,
    const float* __restrict__ v, const float* __restrict__ g,
    const float* __restrict__ beta, float* __restrict__ o) {
  const int b   = blockIdx.x >> 5;
  const int h   = blockIdx.x & 31;
  const int kvh = h >> 1;           // GQA: 32 v-heads share 16 k-heads
  const int tid = threadIdx.x;      // DV column index

  __shared__ float ks[DK], qs[DK];
  float st[DK];
#pragma unroll
  for (int kk = 0; kk < DK; ++kk) st[kk] = 0.f;

  for (int t = 0; t < SS; ++t) {
    const size_t bst = (size_t)(b * SS + t);
    __syncthreads();
    ks[tid] = k[bst * KEY_DIM + kvh * DK + tid];
    qs[tid] = q[bst * KEY_DIM + kvh * DK + tid];
    __syncthreads();

    const float eg = expf(g[bst * NUM_V_HEADS + h]);
    const float bt = beta[bst * NUM_V_HEADS + h];
    const float vt = v[bst * VALUE_DIM + h * DV + tid];

    float kv = 0.f;
#pragma unroll
    for (int kk = 0; kk < DK; ++kk) {
      st[kk] *= eg;
      kv += ks[kk] * st[kk];
    }
    const float delta = (vt - kv) * bt;
    float out = 0.f;
#pragma unroll
    for (int kk = 0; kk < DK; ++kk) {
      st[kk] += ks[kk] * delta;
      out += qs[kk] * st[kk];
    }
    o[bst * VALUE_DIM + h * DV + tid] = out;
  }
}

// ---------------------------------------------------------------------------
// Gated RMSNorm: o *= silu(z); o *= rsqrt(mean(o^2)+eps); o *= gamma -> bf16
// ---------------------------------------------------------------------------
__global__ __launch_bounds__(128) void gated_rmsnorm_kernel(
    const float* __restrict__ o, const __bf16* __restrict__ z,
    const float* __restrict__ gamma, __bf16* __restrict__ onorm) {
  const size_t base = (size_t)blockIdx.x * DV;
  const int tid = threadIdx.x;
  const float zf = (float)z[base + tid];
  const float og = o[base + tid] * (zf / (1.f + expf(-zf)));

  __shared__ float red[128];
  red[tid] = og * og;
  __syncthreads();
  for (int st = 64; st > 0; st >>= 1) {
    if (tid < st) red[tid] += red[tid + st];
    __syncthreads();
  }
  const float inv = rsqrtf(red[0] * (1.f / DV) + 1e-6f);
  onorm[base + tid] = (__bf16)(og * inv * gamma[tid]);
}

// ---------------------------------------------------------------------------
// Host-side orchestration
// ---------------------------------------------------------------------------
namespace {
constexpr size_t HS_BF    = 0;
constexpr size_t WQKV_BF  = HS_BF    + (size_t)BS * H_SIZE * 2;        // hs (M*K)
constexpr size_t WZ_BF    = WQKV_BF  + (size_t)H_SIZE * CONV_DIM * 2;  // w_qkv^T
constexpr size_t WOUT_BF  = WZ_BF    + (size_t)H_SIZE * VALUE_DIM * 2; // w_z^T
constexpr size_t MIXED_BF = WOUT_BF  + (size_t)VALUE_DIM * H_SIZE * 2; // w_out^T
constexpr size_t Z_BF     = MIXED_BF + (size_t)BS * CONV_DIM * 2;
constexpr size_t Q_F      = Z_BF     + (size_t)BS * VALUE_DIM * 2;
constexpr size_t K_F      = Q_F      + (size_t)BS * KEY_DIM * 4;
constexpr size_t V_F      = K_F      + (size_t)BS * KEY_DIM * 4;
constexpr size_t G_F      = V_F      + (size_t)BS * VALUE_DIM * 4;
constexpr size_t BETA_F   = G_F      + (size_t)BS * NUM_V_HEADS * 4;
constexpr size_t O_F      = BETA_F   + (size_t)BS * NUM_V_HEADS * 4;
constexpr size_t ONORM_BF = O_F      + (size_t)BS * VALUE_DIM * 4;
}  // namespace

extern "C" void kernel_launch(void* const* d_in, const int* in_sizes, int n_in,
                              void* d_out, int out_size, void* d_ws, size_t ws_size,
                              hipStream_t stream) {
  (void)in_sizes; (void)n_in; (void)out_size; (void)ws_size;
  const float* hs      = (const float*)d_in[0];
  const float* w_qkv   = (const float*)d_in[1];
  const float* w_z     = (const float*)d_in[2];
  const float* w_b     = (const float*)d_in[3];
  const float* w_a     = (const float*)d_in[4];
  const float* conv_w  = (const float*)d_in[5];
  const float* dt_bias = (const float*)d_in[6];
  const float* A_log   = (const float*)d_in[7];
  const float* gamma   = (const float*)d_in[8];
  const float* w_out   = (const float*)d_in[9];

  char* ws = (char*)d_ws;
  __bf16* hs_bf    = (__bf16*)(ws + HS_BF);
  __bf16* wqkvT_bf = (__bf16*)(ws + WQKV_BF);
  __bf16* wzT_bf   = (__bf16*)(ws + WZ_BF);
  __bf16* woutT_bf = (__bf16*)(ws + WOUT_BF);
  __bf16* mixed_bf = (__bf16*)(ws + MIXED_BF);
  __bf16* z_bf     = (__bf16*)(ws + Z_BF);
  float*  qf       = (float*)(ws + Q_F);
  float*  kf       = (float*)(ws + K_F);
  float*  vf       = (float*)(ws + V_F);
  float*  gf       = (float*)(ws + G_F);
  float*  betaf    = (float*)(ws + BETA_F);
  float*  of       = (float*)(ws + O_F);
  __bf16* onorm_bf = (__bf16*)(ws + ONORM_BF);

  // 1. fp32 -> bf16 converts (weights transposed to N x K for TDM staging)
  {
    int n = BS * H_SIZE;
    cvt_f32_bf16<<<n / 256, 256, 0, stream>>>(hs, hs_bf, n);
    cvt_transpose_bf16<<<dim3(CONV_DIM / 32, H_SIZE / 32), dim3(32, 8), 0, stream>>>(
        w_qkv, wqkvT_bf, H_SIZE, CONV_DIM);
    cvt_transpose_bf16<<<dim3(VALUE_DIM / 32, H_SIZE / 32), dim3(32, 8), 0, stream>>>(
        w_z, wzT_bf, H_SIZE, VALUE_DIM);
    cvt_transpose_bf16<<<dim3(H_SIZE / 32, VALUE_DIM / 32), dim3(32, 8), 0, stream>>>(
        w_out, woutT_bf, VALUE_DIM, H_SIZE);
  }

  // 2. mixed = hs @ w_qkv  (4096 x 8192 x 2048) -> bf16
  gemm_bf16_wmma_tdm<<<dim3(CONV_DIM / 128, BS / 128), 256, 0, stream>>>(
      hs_bf, wqkvT_bf, mixed_bf, BS, CONV_DIM, H_SIZE, 1);

  // 3. z = hs @ w_z  (4096 x 4096 x 2048) -> bf16
  gemm_bf16_wmma_tdm<<<dim3(VALUE_DIM / 128, BS / 128), 256, 0, stream>>>(
      hs_bf, wzT_bf, z_bf, BS, VALUE_DIM, H_SIZE, 1);

  // 4. beta / g gates
  ab_gate_kernel<<<BS, 256, 0, stream>>>(hs, w_b, w_a, dt_bias, A_log, betaf, gf);

  // 5. causal conv + silu + l2norm split into q/k/v
  conv_silu_split_kernel<<<dim3(CONV_DIM / 128, BS), 128, 0, stream>>>(
      mixed_bf, conv_w, qf, kf, vf);

  // 6. sequential gated delta-rule recurrence (state in registers)
  deltanet_recurrence_kernel<<<BB * NUM_V_HEADS, 128, 0, stream>>>(
      qf, kf, vf, gf, betaf, of);

  // 7. gated RMSNorm -> bf16
  gated_rmsnorm_kernel<<<BS * NUM_V_HEADS, 128, 0, stream>>>(of, z_bf, gamma, onorm_bf);

  // 8. final projection: d_out = onorm @ w_out  (4096 x 2048 x 4096) -> fp32
  gemm_bf16_wmma_tdm<<<dim3(H_SIZE / 128, BS / 128), 256, 0, stream>>>(
      onorm_bf, woutT_bf, d_out, BS, H_SIZE, VALUE_DIM, 0);
}
